// FrequencySplitGenerator_28853590294940
// MI455X (gfx1250) — matrix operand order repaired and checked by reading, not compiled, not run on hardware
//
#include <hip/hip_runtime.h>

typedef __attribute__((ext_vector_type(16))) __bf16 v16bf;
typedef __attribute__((ext_vector_type(8)))  __bf16 v8bf;
typedef __attribute__((ext_vector_type(8)))  float  v8f;

#define K_DIM 256
#define NT    32    // real rows staged into LDS per iteration
#define NS    4     // n-range slices (grid.y) for extra grid parallelism

// ---------------------------------------------------------------------------
// helpers
// ---------------------------------------------------------------------------
__device__ __forceinline__ unsigned short f2bf_rne(float f) {
  unsigned int u = __float_as_uint(f);
  u += 0x7FFFu + ((u >> 16) & 1u);
  return (unsigned short)(u >> 16);
}
__device__ __forceinline__ float bf2f(unsigned short h) {
  return __uint_as_float(((unsigned int)h) << 16);
}

// raw LDS byte offset of a __shared__ object (generic -> addrspace(3) cast)
__device__ __forceinline__ unsigned lds_off(const void* p) {
  return (unsigned)(unsigned long long)(const __attribute__((address_space(3))) char*)p;
}

// CDNA5 async copy global -> LDS, 16B per lane, tracked by ASYNCcnt
__device__ __forceinline__ void async_b128(unsigned dst_lds, const void* src) {
  asm volatile("global_load_async_to_lds_b128 %0, %1, off"
               :: "v"(dst_lds), "v"(src) : "memory");
}
#define WAIT_ASYNC(n) asm volatile("s_wait_asynccnt %0" :: "n"(n))

__device__ __forceinline__ void topk_insert(float v, int i,
                                            float& v0, int& i0,
                                            float& v1, int& i1,
                                            float& v2, int& i2) {
  if (v > v2) {
    if (v > v0)       { v2 = v1; i2 = i1; v1 = v0; i1 = i0; v0 = v; i0 = i; }
    else if (v > v1)  { v2 = v1; i2 = i1; v1 = v;  i1 = i; }
    else              { v2 = v;  i2 = i; }
  }
}

// ---------------------------------------------------------------------------
// Kernel 1: L2-normalize each 256-elem row, split into bf16 hi + lo so the
// GEMM runs on the bf16 matrix pipes at ~f32 precision (hi*hi + hi*lo + lo*hi)
// ---------------------------------------------------------------------------
__global__ void __launch_bounds__(256)
normalize_split_kernel(const float* __restrict__ x,
                       unsigned short* __restrict__ hi,
                       unsigned short* __restrict__ lo,
                       int nrows) {
  const int wave = (int)((blockIdx.x * blockDim.x + threadIdx.x) >> 5);
  const int lane = (int)(threadIdx.x & 31);
  if (wave >= nrows) return;

  const float* row = x + (size_t)wave * K_DIM + lane * 8;
  float4 a = *(const float4*)(row);
  float4 b = *(const float4*)(row + 4);

  float v[8] = {a.x, a.y, a.z, a.w, b.x, b.y, b.z, b.w};
  float s = 0.0f;
#pragma unroll
  for (int j = 0; j < 8; ++j) s += v[j] * v[j];
#pragma unroll
  for (int off = 16; off > 0; off >>= 1) s += __shfl_xor(s, off);

  const float inv = 1.0f / fmaxf(sqrtf(s), 1e-12f);

  unsigned short h[8], l[8];
#pragma unroll
  for (int j = 0; j < 8; ++j) {
    float y = v[j] * inv;
    unsigned short hb = f2bf_rne(y);
    h[j] = hb;
    l[j] = f2bf_rne(y - bf2f(hb));
  }

  uint4 hp, lp;
  hp.x = (unsigned)h[0] | ((unsigned)h[1] << 16);
  hp.y = (unsigned)h[2] | ((unsigned)h[3] << 16);
  hp.z = (unsigned)h[4] | ((unsigned)h[5] << 16);
  hp.w = (unsigned)h[6] | ((unsigned)h[7] << 16);
  lp.x = (unsigned)l[0] | ((unsigned)l[1] << 16);
  lp.y = (unsigned)l[2] | ((unsigned)l[3] << 16);
  lp.z = (unsigned)l[4] | ((unsigned)l[5] << 16);
  lp.w = (unsigned)l[6] | ((unsigned)l[7] << 16);
  *(uint4*)(hi + (size_t)wave * K_DIM + lane * 8) = hp;
  *(uint4*)(lo + (size_t)wave * K_DIM + lane * 8) = lp;
}

// ---------------------------------------------------------------------------
// Kernel 2: fused simT-tile GEMM + per-slice top-3.
//  - grid = (Nb/128, NS); block = 256 threads = 8 waves, wave owns 16 base rows
//  - B (bn^T) tiles register-resident per wave for the whole n-loop
//  - A (rn) tiles: 32 rows x 256 K (hi+lo, 32KB) staged in LDS, double-buffered
//    via async global->LDS copies overlapped with the 48 WMMAs per iteration
//  - A fragments register double-buffered; sched_barrier(0) pins the
//    [prefetch-loads | wmma] stage order so s_wait_dscnt never goes to 0
// ---------------------------------------------------------------------------
__global__ void __launch_bounds__(256)
cosine_topk_kernel(const unsigned short* __restrict__ bnHi,
                   const unsigned short* __restrict__ bnLo,
                   const unsigned short* __restrict__ rnHi,
                   const unsigned short* __restrict__ rnLo,
                   float* __restrict__ pv, int* __restrict__ pi,
                   int Nb, int Nr) {
  __shared__ unsigned short smem[2 * 2 * NT * K_DIM];  // [buf][hi|lo][row][col] = 64KB

  const int tid   = (int)threadIdx.x;
  const int lane  = tid & 31;
  const int wv    = tid >> 5;
  const int m0    = ((int)blockIdx.x * 8 + wv) << 4;
  const int slice = (int)blockIdx.y;
  const int sliceRows = Nr / NS;
  const int nBeg  = slice * sliceRows;
  const int nIter = sliceRows / NT;

  const int  c      = lane & 15;
  const bool hiHalf = lane >= 16;

  // --- register-resident B (base) tiles: 8 K-chunks, hi + lo -------------
  v16bf Bhi[8], Blo[8];
  {
    const size_t rb = (size_t)(m0 + c) * K_DIM;
#pragma unroll
    for (int kc = 0; kc < 8; ++kc) {
      const int kb = kc * 32 + (hiHalf ? 16 : 0);  // B: lane<16 K 0-15, else 16-31
      Bhi[kc] = *(const v16bf*)(bnHi + rb + kb);
      Blo[kc] = *(const v16bf*)(bnLo + rb + kb);
    }
  }

  // issue async copy of one NT-row tile (hi+lo) into LDS buffer `buf`
  auto issue_tile = [&](int buf, int n0) {
    unsigned short* sHi = smem + buf * (2 * NT * K_DIM);
    unsigned short* sLo = sHi + NT * K_DIM;
#pragma unroll
    for (int r = 0; r < 4; ++r) {
      const int chunk = tid + r * 256;        // 0..1023 16B-chunks per array
      const int row = chunk >> 5;             // 32 chunks per 512B row
      const int col = (chunk & 31) << 3;      // ushort offset
      async_b128(lds_off(sHi + row * K_DIM + col),
                 rnHi + (size_t)(n0 + row) * K_DIM + col);
      async_b128(lds_off(sLo + row * K_DIM + col),
                 rnLo + (size_t)(n0 + row) * K_DIM + col);
    }
  };

  float v0 = -__builtin_inff(), v1 = -__builtin_inff(), v2 = -__builtin_inff();
  int   i0 = 0, i1 = 0, i2 = 0;

  const int rOff = hiHalf ? 8 : 0;  // D rows: upper half-lanes hold M=r+8
  const int aSeg = hiHalf ? 16 : 0;  // A: lane<16 K 0-15, else 16-31

  issue_tile(0, nBeg);
  for (int it = 0; it < nIter; ++it) {
    const int cur = it & 1;
    if (it + 1 < nIter) {
      issue_tile(cur ^ 1, nBeg + (it + 1) * NT);  // prefetch next tile
      WAIT_ASYNC(8);                               // current tile's 8 copies done
    } else {
      WAIT_ASYNC(0);
    }
    __syncthreads();  // all waves' portions of current tile visible

    const unsigned short* sHi = smem + cur * (2 * NT * K_DIM);
    const unsigned short* sLo = sHi + NT * K_DIM;
    const unsigned short* hRow0 = sHi + c * K_DIM;
    const unsigned short* lRow0 = sLo + c * K_DIM;
    const unsigned short* hRow1 = hRow0 + 16 * K_DIM;
    const unsigned short* lRow1 = lRow0 + 16 * K_DIM;

    // software-pipelined A fragments across the 2 tiles x 8 K-chunks
    v8bf ah0 = *(const v8bf*)(hRow0 + aSeg);
    v8bf ah1 = *(const v8bf*)(hRow0 + aSeg + 8);
    v8bf al0 = *(const v8bf*)(lRow0 + aSeg);
    v8bf al1 = *(const v8bf*)(lRow0 + aSeg + 8);

    v8f acc0 = {}, acc1 = {};
#pragma unroll
    for (int j = 0; j < 16; ++j) {
      v8bf nh0 = {}, nh1 = {}, nl0 = {}, nl1 = {};
      if (j < 15) {
        const int jn = j + 1;
        const unsigned short* hR = (jn & 8) ? hRow1 : hRow0;
        const unsigned short* lR = (jn & 8) ? lRow1 : lRow0;
        const int k0 = (jn & 7) * 32 + aSeg;
        nh0 = *(const v8bf*)(hR + k0);        // prefetch next chunk's fragments
        nh1 = *(const v8bf*)(hR + k0 + 8);
        nl0 = *(const v8bf*)(lR + k0);
        nl1 = *(const v8bf*)(lR + k0 + 8);
      }
      // pin pipeline stage order: prefetch loads may not sink below here,
      // WMMAs may not hoist above -> consumer waits see dscnt<=4, never 0
      __builtin_amdgcn_sched_barrier(0);

      v16bf Ahi = __builtin_shufflevector(ah0, ah1, 0,1,2,3,4,5,6,7,8,9,10,11,12,13,14,15);
      v16bf Alo = __builtin_shufflevector(al0, al1, 0,1,2,3,4,5,6,7,8,9,10,11,12,13,14,15);
      const int kc = j & 7;
      v8f& acc = (j & 8) ? acc1 : acc0;
      acc = __builtin_amdgcn_wmma_f32_16x16x32_bf16(false, Alo, false, Bhi[kc],
                                                    (short)0, acc, false, false);
      acc = __builtin_amdgcn_wmma_f32_16x16x32_bf16(false, Ahi, false, Blo[kc],
                                                    (short)0, acc, false, false);
      acc = __builtin_amdgcn_wmma_f32_16x16x32_bf16(false, Ahi, false, Bhi[kc],
                                                    (short)0, acc, false, false);
      __builtin_amdgcn_sched_barrier(0);

      ah0 = nh0; ah1 = nh1; al0 = nl0; al1 = nl1;
    }

    const int nb = nBeg + it * NT + rOff;
#pragma unroll
    for (int r = 0; r < 8; ++r)
      topk_insert(acc0[r], nb + r, v0, i0, v1, i1, v2, i2);
#pragma unroll
    for (int r = 0; r < 8; ++r)
      topk_insert(acc1[r], nb + 16 + r, v0, i0, v1, i1, v2, i2);

    __syncthreads();  // done reading current buffer before it is overwritten
  }

  // --- merge the two half-lanes (same base column, disjoint real rows) ---
  float ov0 = __shfl_xor(v0, 16), ov1 = __shfl_xor(v1, 16), ov2 = __shfl_xor(v2, 16);
  int   oi0 = __shfl_xor(i0, 16), oi1 = __shfl_xor(i1, 16), oi2 = __shfl_xor(i2, 16);
  topk_insert(ov0, oi0, v0, i0, v1, i1, v2, i2);
  topk_insert(ov1, oi1, v0, i0, v1, i1, v2, i2);
  topk_insert(ov2, oi2, v0, i0, v1, i1, v2, i2);

  if (!hiHalf) {
    const size_t b = ((size_t)slice * Nb + (m0 + c)) * 3;
    pv[b + 0] = v0; pi[b + 0] = i0;
    pv[b + 1] = v1; pi[b + 1] = i1;
    pv[b + 2] = v2; pi[b + 2] = i2;
  }
}

// ---------------------------------------------------------------------------
// Kernel 3: merge the NS per-slice top-3 lists into the final top-3 indices.
// Slices visited in ascending index order -> '>' keeps the earlier index.
// ---------------------------------------------------------------------------
__global__ void __launch_bounds__(256)
topk_merge_kernel(const float* __restrict__ pv, const int* __restrict__ pi,
                  int* __restrict__ out, int Nb) {
  const int m = (int)(blockIdx.x * blockDim.x + threadIdx.x);
  if (m >= Nb) return;
  float v0 = -__builtin_inff(), v1 = -__builtin_inff(), v2 = -__builtin_inff();
  int   i0 = 0, i1 = 0, i2 = 0;
#pragma unroll
  for (int s = 0; s < NS; ++s) {
    const size_t b = ((size_t)s * Nb + m) * 3;
#pragma unroll
    for (int j = 0; j < 3; ++j)
      topk_insert(pv[b + j], pi[b + j], v0, i0, v1, i1, v2, i2);
  }
  out[m * 3 + 0] = i0;
  out[m * 3 + 1] = i1;
  out[m * 3 + 2] = i2;
}

extern "C" void kernel_launch(void* const* d_in, const int* in_sizes, int n_in,
                              void* d_out, int out_size, void* d_ws, size_t ws_size,
                              hipStream_t stream) {
  (void)n_in; (void)out_size; (void)ws_size;
  const float* base = (const float*)d_in[0];  // [Nb, 256] f32
  const float* real = (const float*)d_in[1];  // [Nr, 256] f32
  const int Nb = in_sizes[0] / K_DIM;         // 16384
  const int Nr = in_sizes[1] / K_DIM;         // 8192

  unsigned short* bnHi = (unsigned short*)d_ws;
  unsigned short* bnLo = bnHi + (size_t)Nb * K_DIM;
  unsigned short* rnHi = bnLo + (size_t)Nb * K_DIM;
  unsigned short* rnLo = rnHi + (size_t)Nr * K_DIM;
  float*          pv   = (float*)(rnLo + (size_t)Nr * K_DIM);
  int*            pi   = (int*)(pv + (size_t)NS * Nb * 3);

  normalize_split_kernel<<<(Nb + 7) / 8, 256, 0, stream>>>(base, bnHi, bnLo, Nb);
  normalize_split_kernel<<<(Nr + 7) / 8, 256, 0, stream>>>(real, rnHi, rnLo, Nr);

  dim3 grid(Nb / 128, NS);  // 8 waves/block * 16 rows/wave = 128 base rows/block
  cosine_topk_kernel<<<grid, 256, 0, stream>>>(bnHi, bnLo, rnHi, rnLo,
                                               pv, pi, Nb, Nr);

  topk_merge_kernel<<<(Nb + 255) / 256, 256, 0, stream>>>(pv, pi, (int*)d_out, Nb);
}
